// Channel_attention_80376017977671
// MI455X (gfx1250) — matrix operand order repaired and checked by reading, not compiled, not run on hardware
//
#include <hip/hip_runtime.h>
#include <math.h>

// Fused self-attention (flash-style, two-pass softmax) for
//   q = x.reshape(B,N,C); out = gamma * softmax(q q^T) q + x
// B=2, N=8192, C=64 on gfx1250 (wave32, v_wmma_f32_16x16x32_f16).
//
// Pass A: stream S = Q K^T tiles, track per-row max (register-only, one
//         cross-lane reduction total).
// Pass B: recompute S, p = exp(s - M), O += P V, and the softmax denominator
//         is accumulated as a 5th WMMA accumulator P x ones (no shuffles).
// P is spilled/reloaded through LDS with keys interleaved (pos 2i <-> key i,
// pos 2i+1 <-> key 16+i); V is staged in the same permuted order so the
// K-dim permutation cancels inside the WMMA contraction.

typedef __attribute__((ext_vector_type(16))) _Float16     v16h;
typedef __attribute__((ext_vector_type(8)))  float        v8f;
typedef __attribute__((ext_vector_type(4)))  unsigned int v4u;

#define B_    2
#define N_    8192
#define C_    64
#define KT_   64                  // keys per tile
#define NT_   (N_ / KT_)          // 128 tiles
#define ROWS_PER_BLOCK 128        // 8 waves x 16 query rows
#define NW_   8
#define KSTR  72                  // Kt row stride in halfs (pad for banks)
#define VSTR  72                  // Vt row stride in halfs
#define PSTR  72                  // P2 row stride in halfs

union Frag { v16h h; v4u u[2]; };
union H8   { _Float16 h[8]; v4u u; };
union HP   { _Float16 h[2]; unsigned int u; };

#define WMMA(A, Bf, Cacc) \
    __builtin_amdgcn_wmma_f32_16x16x32_f16(false, (A).h, false, (Bf).h, \
                                           (short)0, (Cacc), false, false)

__global__ __launch_bounds__(256) void
attn_fused_kernel(const float* __restrict__ x,
                  const float* __restrict__ gamma,
                  float* __restrict__ out)
{
    __shared__ __align__(16) _Float16 Kt[KT_][KSTR];       // row-major keys
    __shared__ __align__(16) _Float16 Vt[C_][VSTR];        // transposed, key-permuted
    __shared__ __align__(16) _Float16 P2[NW_][16][PSTR];   // per-wave P spill

    const int tid  = threadIdx.x;
    const int wave = tid >> 5;
    const int lane = tid & 31;
    const int n    = lane & 15;
    const bool lo  = (lane < 16);
    const int kb   = lo ? 0 : 16;     // B-frag K half (channels for S GEMM)
    const int kb1  = lo ? 0 : 16;     // A-frag K base (Q fragments)
    const int ab   = lo ? 0 : 16;     // P A-frag halfs base (positions)
    const int vb   = lo ? 0 : 16;     // V B-frag halfs base (positions)

    const int b     = blockIdx.x / (N_ / ROWS_PER_BLOCK);
    const int rb    = (blockIdx.x % (N_ / ROWS_PER_BLOCK)) * ROWS_PER_BLOCK;
    const int qbase = rb + wave * 16;

    const float* xb = x + (size_t)b * (size_t)(N_ * C_);

    // staging decomposition: each thread converts 2 keys (i, i+16 of a 32-key
    // window) x 8 channels; Vt gets packed b32 stores at positions (2i, 2i+1).
    const int sc0  = (tid & 7) * 8;       // channel base
    const int sip  = tid >> 3;            // 0..31
    const int sw   = sip >> 4;            // 32-key window 0/1
    const int si   = sip & 15;
    const int kA   = 32 * sw + si;
    const int kB   = 32 * sw + 16 + si;
    const int qpos = 32 * sw + 2 * si;    // permuted position of key pair

    // ---- Q A-fragments (16x32 f16), contiguous k = 16h + s, same as Kt B-frags ----
    Frag aq0, aq1;
    {
        const float* qp = xb + (size_t)(qbase + n) * C_;
        #pragma unroll
        for (int e = 0; e < 8; ++e) {
            aq0.h[e]     = (_Float16)qp[kb1 + e];
            aq0.h[8 + e] = (_Float16)qp[kb1 + 8 + e];
            aq1.h[e]     = (_Float16)qp[32 + kb1 + e];
            aq1.h[8 + e] = (_Float16)qp[32 + kb1 + 8 + e];
        }
    }

    // ================= PASS A: row max =================
    float mx[8];
    #pragma unroll
    for (int j = 0; j < 8; ++j) mx[j] = -INFINITY;

    #pragma unroll 1
    for (int kt = 0; kt < NT_; ++kt) {
        __syncthreads();
        {   // stage Kt only
            const float* spA = xb + (size_t)(kt * KT_ + kA) * C_ + sc0;
            const float* spB = xb + (size_t)(kt * KT_ + kB) * C_ + sc0;
            H8 ha, hc;
            #pragma unroll
            for (int i = 0; i < 8; ++i) {
                ha.h[i] = (_Float16)spA[i];
                hc.h[i] = (_Float16)spB[i];
            }
            *(v4u*)&Kt[kA][sc0] = ha.u;
            *(v4u*)&Kt[kB][sc0] = hc.u;
        }
        __syncthreads();

        v8f s[4];
        #pragma unroll
        for (int h = 0; h < 4; ++h) {       // 4 key-halves x (K=64 -> 2 WMMA)
            Frag bk; v8f acc = {};
            bk.u[0] = *(const v4u*)&Kt[16 * h + n][kb];
            bk.u[1] = *(const v4u*)&Kt[16 * h + n][kb + 8];
            acc = WMMA(aq0, bk, acc);
            bk.u[0] = *(const v4u*)&Kt[16 * h + n][32 + kb];
            bk.u[1] = *(const v4u*)&Kt[16 * h + n][32 + kb + 8];
            acc = WMMA(aq1, bk, acc);
            s[h] = acc;
        }
        #pragma unroll
        for (int j = 0; j < 8; ++j)
            mx[j] = fmaxf(mx[j], fmaxf(fmaxf(s[0][j], s[1][j]),
                                       fmaxf(s[2][j], s[3][j])));
    }

    // one cross-lane max reduction total (16-lane halves)
    float M[8];
    #pragma unroll
    for (int j = 0; j < 8; ++j) {
        float v = mx[j];
        #pragma unroll
        for (int off = 1; off < 16; off <<= 1)
            v = fmaxf(v, __shfl_xor(v, off, 32));
        M[j] = v;
    }

    // ================= PASS B: O = sum exp(S-M) V ; l via P x ones =========
    v8f oacc[4] = {v8f{}, v8f{}, v8f{}, v8f{}};
    v8f osum = {};
    Frag ones;
    #pragma unroll
    for (int e = 0; e < 16; ++e) ones.h[e] = (_Float16)1.0f;

    _Float16 (*Pw)[PSTR] = P2[wave];

    #pragma unroll 1
    for (int kt = 0; kt < NT_; ++kt) {
        __syncthreads();
        {   // stage Kt (row-major) + Vt (transposed, key-permuted, b32 packed)
            const float* spA = xb + (size_t)(kt * KT_ + kA) * C_ + sc0;
            const float* spB = xb + (size_t)(kt * KT_ + kB) * C_ + sc0;
            H8 ha, hc;
            #pragma unroll
            for (int i = 0; i < 8; ++i) {
                ha.h[i] = (_Float16)spA[i];
                hc.h[i] = (_Float16)spB[i];
            }
            *(v4u*)&Kt[kA][sc0] = ha.u;
            *(v4u*)&Kt[kB][sc0] = hc.u;
            #pragma unroll
            for (int i = 0; i < 8; ++i) {
                HP pr; pr.h[0] = ha.h[i]; pr.h[1] = hc.h[i];
                *(unsigned int*)&Vt[sc0 + i][qpos] = pr.u;
            }
        }
        __syncthreads();

        v8f s[4];
        #pragma unroll
        for (int h = 0; h < 4; ++h) {
            Frag bk; v8f acc = {};
            bk.u[0] = *(const v4u*)&Kt[16 * h + n][kb];
            bk.u[1] = *(const v4u*)&Kt[16 * h + n][kb + 8];
            acc = WMMA(aq0, bk, acc);
            bk.u[0] = *(const v4u*)&Kt[16 * h + n][32 + kb];
            bk.u[1] = *(const v4u*)&Kt[16 * h + n][32 + kb + 8];
            acc = WMMA(aq1, bk, acc);
            s[h] = acc;
        }

        // p = exp(s - M); packed b32 spill: dword i of a row = {key i, key 16+i}
        #pragma unroll
        for (int j = 0; j < 8; ++j) {
            const int r = j + (lo ? 0 : 8);
            const float mj = M[j];
            HP w0, w1;
            w0.h[0] = (_Float16)__expf(s[0][j] - mj);   // window0 key n
            w0.h[1] = (_Float16)__expf(s[1][j] - mj);   // window0 key 16+n
            w1.h[0] = (_Float16)__expf(s[2][j] - mj);   // window1 key n
            w1.h[1] = (_Float16)__expf(s[3][j] - mj);   // window1 key 16+n
            *(unsigned int*)&Pw[r][2 * n]      = w0.u;
            *(unsigned int*)&Pw[r][32 + 2 * n] = w1.u;
        }

        // P A-fragments: one contiguous 32B run per lane per window
        Frag ap0, ap1;
        ap0.u[0] = *(const v4u*)&Pw[n][ab];
        ap0.u[1] = *(const v4u*)&Pw[n][ab + 8];
        ap1.u[0] = *(const v4u*)&Pw[n][32 + ab];
        ap1.u[1] = *(const v4u*)&Pw[n][32 + ab + 8];

        // O += P V (key-permuted V matches P's K-dim permutation)
        #pragma unroll
        for (int t = 0; t < 4; ++t) {
            Frag bv;
            const int c = t * 16 + n;
            bv.u[0] = *(const v4u*)&Vt[c][vb];
            bv.u[1] = *(const v4u*)&Vt[c][vb + 8];
            oacc[t] = WMMA(ap0, bv, oacc[t]);
            bv.u[0] = *(const v4u*)&Vt[c][32 + vb];
            bv.u[1] = *(const v4u*)&Vt[c][32 + vb + 8];
            oacc[t] = WMMA(ap1, bv, oacc[t]);
        }
        // denominator: l += P x ones (every column = row sum, right lanes)
        osum = WMMA(ap0, ones, osum);
        osum = WMMA(ap1, ones, osum);
    }

    // ================= epilogue: out = gamma * (O / l) + x =================
    const float g = gamma[0];
    float* ob = out + (size_t)b * (size_t)(N_ * C_);
    #pragma unroll
    for (int j = 0; j < 8; ++j) {
        const int r = j + (lo ? 0 : 8);
        const float inv = 1.0f / osum[j];
        const size_t base = (size_t)(qbase + r) * C_;
        ob[base + n]      = g * (oacc[0][j] * inv) + xb[base + n];
        ob[base + 16 + n] = g * (oacc[1][j] * inv) + xb[base + 16 + n];
        ob[base + 32 + n] = g * (oacc[2][j] * inv) + xb[base + 32 + n];
        ob[base + 48 + n] = g * (oacc[3][j] * inv) + xb[base + 48 + n];
    }
}

extern "C" void kernel_launch(void* const* d_in, const int* in_sizes, int n_in,
                              void* d_out, int out_size, void* d_ws, size_t ws_size,
                              hipStream_t stream)
{
    (void)in_sizes; (void)n_in; (void)out_size; (void)d_ws; (void)ws_size;
    const float* x     = (const float*)d_in[0];
    const float* gamma = (const float*)d_in[1];
    float*       out   = (float*)d_out;

    dim3 grid(B_ * (N_ / ROWS_PER_BLOCK));   // 128 blocks
    dim3 block(256);                          // 8 wave32 waves
    attn_fused_kernel<<<grid, block, 0, stream>>>(x, gamma, out);
}
